// LateInteraction_7206955122754
// MI455X (gfx1250) — matrix operand order repaired and checked by reading, not compiled, not run on hardware
//
#include <hip/hip_runtime.h>
#include <stdint.h>

#define ALPHA 0.5f
#define HDIM 768
#define NCHUNK (HDIM / 32)   // 24 K-slices of 32 for bf16 WMMA
#define KSPLIT 4             // k-range splits per (b, 64-query block)
#define AP 776               // LDS A row stride in halfs (768 + 8 pad)
#define BP 24                // LDS B row stride in halfs (16 + 8 pad)
#define PAIRH (HDIM * BP)    // halfs per B buffer (one 16-key tile)

typedef __attribute__((ext_vector_type(16))) __bf16 v16bf;
typedef __attribute__((ext_vector_type(8)))  float  v8f;
typedef int v4i __attribute__((vector_size(16)));
typedef __attribute__((address_space(1))) v4i* gv4i_p;   // global b128
typedef __attribute__((address_space(3))) v4i* lv4i_p;   // LDS b128

union FragU { uint4 u[2]; v16bf v; };

#if __has_builtin(__builtin_amdgcn_global_load_async_to_lds_b128)
#define HAVE_ASYNC 1
#else
#define HAVE_ASYNC 0
#endif

#if HAVE_ASYNC
static __device__ __forceinline__ void async_b128(const void* g, void* l) {
    __builtin_amdgcn_global_load_async_to_lds_b128((gv4i_p)g, (lv4i_p)l, 0, 0);
}
static __device__ __forceinline__ void wait_async_all() {
#if __has_builtin(__builtin_amdgcn_s_wait_asynccnt)
    __builtin_amdgcn_s_wait_asynccnt(0);
#else
    asm volatile("s_wait_asynccnt 0x0" ::: "memory");
#endif
}
#endif

static __device__ __forceinline__ unsigned short f2bf(float f) {
    union { float f; unsigned int u; } c; c.f = f;
    unsigned int u = c.u;
    unsigned int r = u + 0x7FFFu + ((u >> 16) & 1u);  // round-to-nearest-even
    return (unsigned short)(r >> 16);
}

// ---------------------------------------------------------------------------
// Kernel 1: L2-normalize rows, convert to bf16.
//   rows [0, B*S)      -> qn  [B,S,H] row-major (A-fragment friendly)
//   rows [B*S, 2*B*S)  -> knt [B][S/16][H][16] tile-packed (contiguous B tiles)
// ---------------------------------------------------------------------------
__global__ void __launch_bounds__(128)
li_normalize_kernel(const float* __restrict__ q, const float* __restrict__ k,
                    unsigned short* __restrict__ qn, unsigned short* __restrict__ knt,
                    int B, int S) {
    const int row   = blockIdx.x;
    const int total = B * S;
    const bool isK  = row >= total;
    const int  r    = isK ? row - total : row;
    const float* src = (isK ? k : q) + (size_t)r * HDIM;
    const int tid = threadIdx.x;            // 128 threads, 6 elems each

    float vals[6];
    float ss = 0.f;
#pragma unroll
    for (int i = 0; i < 6; ++i) {
        float v = src[tid + i * 128];
        vals[i] = v;
        ss += v * v;
    }
#pragma unroll
    for (int off = 16; off >= 1; off >>= 1) ss += __shfl_xor(ss, off, 32);
    __shared__ float sred[4];
    if ((tid & 31) == 0) sred[tid >> 5] = ss;
    __syncthreads();
    float tot = sred[0] + sred[1] + sred[2] + sred[3];
    float inv = 1.0f / fmaxf(sqrtf(tot), 1e-12f);

    if (!isK) {
        unsigned short* dst = qn + (size_t)r * HDIM;
#pragma unroll
        for (int i = 0; i < 6; ++i) dst[tid + i * 128] = f2bf(vals[i] * inv);
    } else {
        const int b = r / S, s = r - b * S;
        unsigned short* dst = knt + (size_t)b * S * HDIM
                            + (size_t)(s >> 4) * (HDIM * 16) + (s & 15);
#pragma unroll
        for (int i = 0; i < 6; ++i) {
            int h = tid + i * 128;
            dst[(size_t)h * 16] = f2bf(vals[i] * inv);
        }
    }
}

// ---------------------------------------------------------------------------
// Kernel 2: fused sim-GEMM (bf16 WMMA) + decay + masked softmax-ratio.
// Block = 128 threads (4 waves): 64 queries (wave w -> rows qbase+16w..),
// strip of (S/16)/KSPLIT key tiles processed as PAIRS with explicit
// fragment software pipelining; B pairs double-buffered via async global->LDS.
// Emits partial (N, D) per query row into partials[B][S][KSPLIT][2].
// ---------------------------------------------------------------------------
__global__ void __launch_bounds__(128)
li_attn_kernel(const unsigned short* __restrict__ qn,
               const unsigned short* __restrict__ knt,
               const float* __restrict__ q_mask,
               const float* __restrict__ k_mask,
               const float* __restrict__ temp_p,
               float* __restrict__ partials,
               int S) {
    __shared__ __attribute__((aligned(16))) unsigned short As[64 * AP];
    __shared__ __attribute__((aligned(16))) unsigned short Bs[4 * PAIRH];

    const int qblocks = S >> 6;
    const int ks  = blockIdx.x % KSPLIT;
    const int qb  = (blockIdx.x / KSPLIT) % qblocks;
    const int b   = blockIdx.x / (KSPLIT * qblocks);
    const int qbase = qb << 6;
    const int tid  = threadIdx.x;
    const int lane = tid & 31, wave = tid >> 5;
    const int half = lane >> 4, l15 = lane & 15;
    const int ktiles = (S >> 4) / KSPLIT;   // 32
    const int npairs = ktiles >> 1;         // 16
    const int kt0 = ks * ktiles;
    const float invT = 1.0f / temp_p[0];

    const uint4* q4  = (const uint4*)(qn + ((size_t)b * S + qbase) * HDIM);
    const uint4* kt4 = (const uint4*)(knt + (size_t)b * S * HDIM);

    // ---- stage A (64 query rows) + first B pair ----
#if HAVE_ASYNC
#pragma unroll
    for (int i = 0; i < 48; ++i) {
        int g = tid + i * 128;              // 6144 uint4 total
        int row = g / 96, col = g - row * 96;
        async_b128(&q4[row * 96 + col], As + row * AP + col * 8);
    }
    {
        size_t tb = (size_t)kt0 * 1536;     // 2 tiles = 3072 uint4, contiguous
#pragma unroll
        for (int i = 0; i < 24; ++i) {
            int g = tid + i * 128;
            int tile = (i >= 12) ? 1 : 0;
            int g2 = g - tile * 1536;
            int h = g2 >> 1, part = g2 & 1;
            async_b128(&kt4[tb + g], Bs + tile * PAIRH + h * BP + part * 8);
        }
    }
    wait_async_all();
#else
#pragma unroll
    for (int i = 0; i < 48; ++i) {
        int g = tid + i * 128;
        int row = g / 96, col = g - row * 96;
        uint4 v = q4[row * 96 + col];
        *(uint4*)(As + row * AP + col * 8) = v;
    }
    {
        size_t tb = (size_t)kt0 * 1536;
#pragma unroll
        for (int i = 0; i < 24; ++i) {
            int g = tid + i * 128;
            int tile = (i >= 12) ? 1 : 0;
            int g2 = g - tile * 1536;
            int h = g2 >> 1, part = g2 & 1;
            uint4 v = kt4[tb + g];
            *(uint4*)(Bs + tile * PAIRH + h * BP + part * 8) = v;
        }
    }
#endif
    __syncthreads();

    float qm[8];
#pragma unroll
    for (int r = 0; r < 8; ++r)
        qm[r] = q_mask[(size_t)b * S + qbase + wave * 16 + half * 8 + r];

    float accN[8], accD[8];
#pragma unroll
    for (int r = 0; r < 8; ++r) { accN[r] = 0.f; accD[r] = 0.f; }

    const unsigned short* Arow = As + (wave * 16 + l15) * AP;

    for (int p = 0; p < npairs; ++p) {
        const int par = p & 1;
        const int kbase = (kt0 + 2 * p) << 4;

#if !HAVE_ASYNC
        uint4 st[24];
#endif
        if (p + 1 < npairs) {
            size_t tb = (size_t)(kt0 + 2 * (p + 1)) * 1536;
#if HAVE_ASYNC
            unsigned short* Bn = Bs + (par ^ 1) * (2 * PAIRH);
#pragma unroll
            for (int i = 0; i < 24; ++i) {
                int g = tid + i * 128;
                int tile = (i >= 12) ? 1 : 0;
                int g2 = g - tile * 1536;
                int h = g2 >> 1, part = g2 & 1;
                async_b128(&kt4[tb + g], Bn + tile * PAIRH + h * BP + part * 8);
            }
#else
#pragma unroll
            for (int i = 0; i < 24; ++i) st[i] = kt4[tb + tid + i * 128];
#endif
        }

        const float km0 = k_mask[(size_t)b * S + kbase + l15];
        const float km1 = k_mask[(size_t)b * S + kbase + 16 + l15];
        const unsigned short* B0 = Bs + par * (2 * PAIRH);
        const unsigned short* B1 = B0 + PAIRH;

        v8f c0 = {}, c1 = {};
        // explicit software pipeline: fragments for chunk ch+1 are loaded
        // before the two WMMAs of chunk ch execute.
        FragU A_, B0_, B1_;
        {
            const int boff = l15;
            A_.u[0]  = *(const uint4*)(Arow + half * 8);
            A_.u[1]  = *(const uint4*)(Arow + 16 + half * 8);
#pragma unroll
            for (int s = 0; s < 16; ++s) {
                const int kk = (s < 8) ? (half * 8 + s) : (16 + half * 8 + s - 8);
                B0_.v[s] = __builtin_bit_cast(__bf16, B0[kk * BP + boff]);
                B1_.v[s] = __builtin_bit_cast(__bf16, B1[kk * BP + boff]);
            }
        }
#pragma unroll
        for (int ch = 0; ch < NCHUNK; ++ch) {
            FragU An, B0n, B1n;
            if (ch + 1 < NCHUNK) {
                const int h0 = (ch + 1) * 32;
                const int boff = l15;
                An.u[0]  = *(const uint4*)(Arow + h0 + half * 8);
                An.u[1]  = *(const uint4*)(Arow + h0 + 16 + half * 8);
#pragma unroll
                for (int s = 0; s < 16; ++s) {
                    const int kk = h0 + ((s < 8) ? (half * 8 + s) : (16 + half * 8 + s - 8));
                    B0n.v[s] = __builtin_bit_cast(__bf16, B0[kk * BP + boff]);
                    B1n.v[s] = __builtin_bit_cast(__bf16, B1[kk * BP + boff]);
                }
            }
            c0 = __builtin_amdgcn_wmma_f32_16x16x32_bf16(
                    false, A_.v, false, B0_.v, (short)0, c0, false, false);
            c1 = __builtin_amdgcn_wmma_f32_16x16x32_bf16(
                    false, A_.v, false, B1_.v, (short)0, c1, false, false);
            if (ch + 1 < NCHUNK) { A_ = An; B0_ = B0n; B1_ = B1n; }
        }

        // epilogue: per-lane entries (M = wave*16 + half*8 + r, N = l15)
        const int j0 = kbase + l15;
#pragma unroll
        for (int r = 0; r < 8; ++r) {
            const int i = qbase + wave * 16 + half * 8 + r;
            int d0 = i - j0;        d0 = d0 < 0 ? -d0 : d0;
            int d1 = i - j0 - 16;   d1 = d1 < 0 ? -d1 : d1;
            const float sim0 = c0[r], sim1 = c1[r];
            const float e0 = qm[r] * km0 * __expf(sim0 * __expf(-ALPHA * (float)d0) * invT);
            const float e1 = qm[r] * km1 * __expf(sim1 * __expf(-ALPHA * (float)d1) * invT);
            accN[r] += e0 * sim0 + e1 * sim1;
            accD[r] += e0 + e1;
        }

#if !HAVE_ASYNC
        if (p + 1 < npairs) {
            unsigned short* Bn = Bs + (par ^ 1) * (2 * PAIRH);
#pragma unroll
            for (int i = 0; i < 24; ++i) {
                int g = tid + i * 128;
                int tile = (i >= 12) ? 1 : 0;
                int g2 = g - tile * 1536;
                int h = g2 >> 1, part = g2 & 1;
                *(uint4*)(Bn + tile * PAIRH + h * BP + part * 8) = st[i];
            }
        }
#endif
#if HAVE_ASYNC
        if (p + 1 < npairs) wait_async_all();
#endif
        __syncthreads();
    }

    // reduce across the 16 lanes of each half (N columns)
#pragma unroll
    for (int r = 0; r < 8; ++r) {
#pragma unroll
        for (int off = 1; off < 16; off <<= 1) {
            accN[r] += __shfl_xor(accN[r], off, 32);
            accD[r] += __shfl_xor(accD[r], off, 32);
        }
    }
    if (l15 == 0) {   // lane 0 (rows +0..7) and lane 16 (rows +8..15)
#pragma unroll
        for (int r = 0; r < 8; ++r) {
            int row = qbase + wave * 16 + half * 8 + r;
            size_t pidx = ((size_t)(b * S + row) * KSPLIT + ks) * 2;
            partials[pidx + 0] = accN[r];
            partials[pidx + 1] = accD[r];
        }
    }
}

// ---------------------------------------------------------------------------
// Kernel 3: combine KSPLIT partials, score rows, reduce per batch.
// ---------------------------------------------------------------------------
__global__ void __launch_bounds__(256)
li_finalize_kernel(const float* __restrict__ partials,
                   const float* __restrict__ q_mask,
                   float* __restrict__ out, int S) {
    const int b = blockIdx.x;
    const int tid = threadIdx.x;
    float s = 0.f, m = 0.f;
    for (int row = tid; row < S; row += 256) {
        float N = 0.f, D = 0.f;
#pragma unroll
        for (int p = 0; p < KSPLIT; ++p) {
            size_t idx = ((size_t)(b * S + row) * KSPLIT + p) * 2;
            N += partials[idx + 0];
            D += partials[idx + 1];
        }
        float qmr = q_mask[(size_t)b * S + row];
        s += qmr * N / fmaxf(D, 1e-20f);
        m += qmr;
    }
#pragma unroll
    for (int off = 16; off >= 1; off >>= 1) {
        s += __shfl_xor(s, off, 32);
        m += __shfl_xor(m, off, 32);
    }
    __shared__ float ls[8], lm[8];
    if ((tid & 31) == 0) { ls[tid >> 5] = s; lm[tid >> 5] = m; }
    __syncthreads();
    if (tid == 0) {
        float S_ = 0.f, M_ = 0.f;
        for (int w = 0; w < 8; ++w) { S_ += ls[w]; M_ += lm[w]; }
        out[b] = S_ / fmaxf(M_, 1.0f);
    }
}

// ---------------------------------------------------------------------------
extern "C" void kernel_launch(void* const* d_in, const int* in_sizes, int n_in,
                              void* d_out, int out_size, void* d_ws, size_t ws_size,
                              hipStream_t stream) {
    const float* q      = (const float*)d_in[0];   // [B,S,H]
    const float* k      = (const float*)d_in[1];   // [B,S,H]
    const float* q_mask = (const float*)d_in[2];   // [B,S]
    const float* k_mask = (const float*)d_in[3];   // [B,S]
    const float* temp_p = (const float*)d_in[4];   // scalar

    const int B = out_size;                        // 8
    const int S = in_sizes[2] / B;                 // 2048

    const size_t nQK = (size_t)B * S * HDIM;
    unsigned short* qn  = (unsigned short*)d_ws;        // bf16 [B,S,H]
    unsigned short* knt = qn + nQK;                     // bf16 [B][S/16][H][16]
    float* partials     = (float*)(knt + nQK);          // f32  [B,S,KSPLIT,2]
    float* out          = (float*)d_out;

    li_normalize_kernel<<<dim3(2 * B * S), dim3(128), 0, stream>>>(
        q, k, qn, knt, B, S);

    li_attn_kernel<<<dim3(B * (S / 64) * KSPLIT), dim3(128), 0, stream>>>(
        qn, knt, q_mask, k_mask, temp_p, partials, S);

    li_finalize_kernel<<<dim3(B), dim3(256), 0, stream>>>(
        partials, q_mask, out, S);
}
